// PPGML_Proxy_Anchor_82317343195173
// MI455X (gfx1250) — matrix operand order repaired and checked by reading, not compiled, not run on hardware
//
#include <hip/hip_runtime.h>

typedef __attribute__((ext_vector_type(16))) _Float16 v16h;
typedef __attribute__((ext_vector_type(8)))  _Float16 v8h;
typedef __attribute__((ext_vector_type(8)))  float    v8f;
typedef __attribute__((ext_vector_type(4)))  unsigned int u32x4;
typedef __attribute__((ext_vector_type(8)))  int i32x8;
typedef __attribute__((ext_vector_type(4)))  int i32x4;

#define B_N 512
#define E_N 1024
#define C_N 512
#define KNN 5
#define INF_F 1.0e10f
#define OFFSET_F 0.5f
#define NEV 8
#define EIG_IT 40

// ---------------- generic helpers ----------------

__device__ __forceinline__ float blk_reduce(float v, float* red) {
    int t = threadIdx.x;
    red[t] = v; __syncthreads();
    for (int o = blockDim.x >> 1; o > 0; o >>= 1) {
        if (t < o) red[t] += red[t + o];
        __syncthreads();
    }
    float r = red[0]; __syncthreads();
    return r;
}

__global__ void k_fillf(float* p, int n, float v) {
    int i = blockIdx.x * blockDim.x + threadIdx.x;
    if (i < n) p[i] = v;
}

// row L2 norms + f16 staging (raw and normalized copies)
__global__ void k_rownorm(const float* __restrict__ src, int ncols, float* sq_out,
                          _Float16* raw16, _Float16* norm16) {
    __shared__ float red[256];
    int row = blockIdx.x;
    const float* r = src + (size_t)row * ncols;
    float s = 0.f;
    for (int c = threadIdx.x; c < ncols; c += blockDim.x) { float v = r[c]; s += v * v; }
    float n2 = blk_reduce(s, red);
    float inv = rsqrtf(fmaxf(n2, 1e-30f));
    if (sq_out && threadIdx.x == 0) sq_out[row] = n2;
    for (int c = threadIdx.x; c < ncols; c += blockDim.x) {
        float v = r[c];
        if (raw16)  raw16[(size_t)row * ncols + c]  = (_Float16)v;
        if (norm16) norm16[(size_t)row * ncols + c] = (_Float16)(v * inv);
    }
}

// ---------------- WMMA GEMM: C[M,N] = A[M,K] * B[N,K]^T (f16 in, f32 out) ----
// Block = 8 waves sharing one 16xK B panel staged into LDS by the TDM.
// Wave w computes the 16x16 tile (mg*8+w, nt) with v_wmma_f32_16x16x32_f16.
__global__ void __launch_bounds__(256) k_gemm16(const _Float16* __restrict__ A,
                                                const _Float16* __restrict__ Bm,
                                                float* __restrict__ Cm,
                                                int M, int N, int K) {
    __shared__ _Float16 smemB[16 * E_N]; // 32 KB panel (K <= 1024)
    int wv = threadIdx.x >> 5, lane = threadIdx.x & 31;
    int ntile = N >> 4;
    int nt = blockIdx.x % ntile;
    int mg = blockIdx.x / ntile;
    int mt = mg * 8 + wv;
    int half = lane >> 4, r = lane & 15;

#if defined(__HIP_DEVICE_COMPILE__) && \
    __has_builtin(__builtin_amdgcn_tensor_load_to_lds) && \
    __has_builtin(__builtin_amdgcn_s_wait_tensorcnt)
    if (threadIdx.x == 0) {
        // Tensor DMA descriptor (ISA 8.3/8.4): 2D tile, 16 rows x K cols, 2B elems
        unsigned long long gaddr =
            (unsigned long long)(uintptr_t)(Bm + (size_t)nt * 16 * K);
        unsigned ldsoff = (unsigned)(uintptr_t)&smemB[0]; // low 32b = LDS offset
        u32x4 g0;
        g0[0] = 1u;                                      // count=1, user mode
        g0[1] = ldsoff;                                  // lds_addr
        g0[2] = (unsigned)(gaddr & 0xffffffffu);         // global_addr[31:0]
        g0[3] = (unsigned)((gaddr >> 32) & 0x1ffffffu)   // global_addr[56:32]
              | (2u << 30);                              // type=2 (image)
        unsigned td0 = (unsigned)K;          // tensor_dim0 (elements)
        unsigned td1 = (unsigned)(M > N ? M : N); // tensor rows (OOB bound)
        unsigned tl0 = (unsigned)K;          // tile_dim0
        unsigned tl1 = 16u;                  // tile_dim1 (rows)
        unsigned long long s0 = (unsigned long long)K;   // tensor_dim0_stride
        i32x8 g1;
        g1[0] = (int)(1u << 16);                                  // data_size=2B
        g1[1] = (int)((td0 & 0xffffu) << 16);                     // dim0 lo16 @48
        g1[2] = (int)(((td0 >> 16) & 0xffffu) | ((td1 & 0xffffu) << 16));
        g1[3] = (int)(((td1 >> 16) & 0xffffu) | ((tl0 & 0xffffu) << 16));
        g1[4] = (int)(tl1 & 0xffffu);                             // tile_dim1
        g1[5] = (int)(s0 & 0xffffffffu);
        g1[6] = (int)((s0 >> 32) & 0xffffu);
        g1[7] = 0;
        i32x4 z4 = {0, 0, 0, 0};
#if __clang_major__ >= 23
        i32x8 z8 = {0, 0, 0, 0, 0, 0, 0, 0};
        __builtin_amdgcn_tensor_load_to_lds(g0, g1, z4, z4, z8, 0);
#else
        __builtin_amdgcn_tensor_load_to_lds(g0, g1, z4, z4, 0);
#endif
        __builtin_amdgcn_s_wait_tensorcnt(0);
    }
#else
    for (int idx = threadIdx.x; idx < 16 * K; idx += blockDim.x)
        smemB[idx] = Bm[(size_t)nt * 16 * K + idx];
#endif
    __syncthreads();

    const _Float16* arow = A + (size_t)(mt * 16 + r) * K;
    const _Float16* bl = smemB + (size_t)r * K;
    v8f acc = {};
    for (int kb = 0; kb < K; kb += 32) {
        if (kb + 64 < K)
            __builtin_prefetch((const void*)(arow + kb + 64), 0, 1);
        // A fragment: lane holds M=r, contiguous 16 K values (matches B)
        v16h a = *(const v16h*)(arow + kb + half * 16);
        // B fragment from LDS: lane holds N=r, contiguous 16 K values
        v16h b = *(const v16h*)(bl + kb + half * 16);
        acc = __builtin_amdgcn_wmma_f32_16x16x32_f16(
            false, a, false, b, (short)0, acc, false, false);
    }
    float* crow = Cm + (size_t)(mt * 16 + half * 8) * N + nt * 16 + r;
#pragma unroll
    for (int j = 0; j < 8; j++) crow[(size_t)j * N] = acc[j];
}

// ---------------- isomap graph ----------------

__global__ void k_dist(const float* gram, const float* sq, float* dist, int n) {
    int idx = blockIdx.x * blockDim.x + threadIdx.x;
    if (idx >= n * n) return;
    int i = idx / n, j = idx % n;
    float d2 = sq[i] + sq[j] - 2.f * gram[idx];
    dist[idx] = sqrtf(fmaxf(d2, 0.f));
}

__global__ void k_knn(const float* dist, float* adj, int n) {
    int i = blockIdx.x * blockDim.x + threadIdx.x;
    if (i >= n) return;
    int chosen[KNN];
    for (int k = 0; k < KNN; k++) {
        float best = 2.f * INF_F; int bj = 0;
        for (int j = 0; j < n; j++) {
            if (j == i) continue;
            bool used = false;
            for (int p = 0; p < k; p++) if (chosen[p] == j) used = true;
            if (used) continue;
            float d = dist[(size_t)i * n + j];
            if (d < best) { best = d; bj = j; }
        }
        chosen[k] = bj;
        adj[(size_t)i * n + bj] = best;
    }
}

__global__ void k_sym(float* adj, int n) {
    int idx = blockIdx.x * blockDim.x + threadIdx.x;
    if (idx >= n * n) return;
    int i = idx / n, j = idx % n;
    if (i == j) { adj[idx] = 0.f; return; }
    float a = adj[idx], b = adj[(size_t)j * n + i];
    adj[idx] = fminf(a, b);
}

__global__ void k_fw_step(float* __restrict__ g, int n, int k) {
    int idx = blockIdx.x * blockDim.x + threadIdx.x;
    if (idx >= n * n) return;
    int i = idx / n, j = idx % n;
    float via = g[(size_t)i * n + k] + g[(size_t)k * n + j];
    if (via < g[idx]) g[idx] = via;
}

__global__ void k_gmax(const float* g, int n2, float* out) {
    __shared__ float red[1024];
    float m = 0.f;
    for (int idx = threadIdx.x; idx < n2; idx += blockDim.x) {
        float v = g[idx];
        if (v < 0.5f * INF_F && v > m) m = v;
    }
    red[threadIdx.x] = m; __syncthreads();
    for (int o = blockDim.x >> 1; o > 0; o >>= 1) {
        if (threadIdx.x < o) red[threadIdx.x] = fmaxf(red[threadIdx.x], red[threadIdx.x + o]);
        __syncthreads();
    }
    if (threadIdx.x == 0) out[0] = red[0];
}

__global__ void k_gfix(float* g, int n2, const float* gmax) {
    int idx = blockIdx.x * blockDim.x + threadIdx.x;
    if (idx >= n2) return;
    float v = g[idx];
    g[idx] = (v < 0.5f * INF_F) ? v : gmax[0];
}

__global__ void k_kmrow(const float* g, int n, float* rm) {
    __shared__ float red[256];
    int i = blockIdx.x;
    float s = 0.f;
    for (int j = threadIdx.x; j < n; j += blockDim.x) {
        float v = g[(size_t)i * n + j];
        s += -0.5f * v * v;
    }
    float t = blk_reduce(s, red);
    if (threadIdx.x == 0) rm[i] = t / n;
}

__global__ void k_meanvec(const float* v, int n, float* out) {
    __shared__ float red[512];
    float x = (threadIdx.x < n) ? v[threadIdx.x] : 0.f;
    float s = blk_reduce(x, red);
    if (threadIdx.x == 0) out[0] = s / n;
}

__global__ void k_kmat(const float* g, const float* rm, const float* mt,
                       float* kmat, int n) {
    int idx = blockIdx.x * blockDim.x + threadIdx.x;
    if (idx >= n * n) return;
    int i = idx / n, j = idx % n;
    float v = g[idx];
    kmat[idx] = -0.5f * v * v - rm[i] - rm[j] + mt[0];
}

// ---------------- subspace iteration for top-NEV eigenpairs ----------------

__global__ void k_vinit(float* V, int n, int nev) {
    int idx = blockIdx.x * blockDim.x + threadIdx.x;
    if (idx >= n * nev) return;
    int i = idx / nev, j = idx % nev;
    float h = sinf((float)(i + 1) * 12.9898f + (float)(j + 1) * 78.233f) * 43758.5453f;
    V[idx] = h - floorf(h) - 0.5f;
}

__global__ void k_gemmKV(const float* __restrict__ kmat, const float* __restrict__ V,
                         float* __restrict__ W, int n) {
    int idx = blockIdx.x * blockDim.x + threadIdx.x;
    if (idx >= n * NEV) return;
    int i = idx / NEV, j = idx % NEV;
    const float* row = kmat + (size_t)i * n;
    float s = 0.f;
    for (int t = 0; t < n; t++) s += row[t] * V[t * NEV + j];
    W[idx] = s;
}

__global__ void k_mgs(const float* Wm, float* Vm, int n) {
    __shared__ float red[512];
    int i = threadIdx.x;
    float w[NEV];
    for (int j = 0; j < NEV; j++) w[j] = Wm[i * NEV + j];
    for (int j = 0; j < NEV; j++) {
        float d = blk_reduce(w[j] * w[j], red);
        float inv = rsqrtf(fmaxf(d, 1e-20f));
        w[j] *= inv;
        for (int l = j + 1; l < NEV; l++) {
            float dd = blk_reduce(w[j] * w[l], red);
            w[l] -= dd * w[j];
        }
    }
    for (int j = 0; j < NEV; j++) Vm[i * NEV + j] = w[j];
}

__global__ void k_rayleigh(const float* Vm, const float* Wm, float* lam, int n) {
    __shared__ float red[512];
    int i = threadIdx.x;
    for (int j = 0; j < NEV; j++) {
        float s = blk_reduce(Vm[i * NEV + j] * Wm[i * NEV + j], red);
        if (i == 0) lam[j] = s;
    }
}

__global__ void k_sortcols(float* Vm, float* lam, int n) {
    __shared__ int ord[NEV];
    if (threadIdx.x == 0) {
        float l[NEV]; int used[NEV];
        for (int j = 0; j < NEV; j++) { l[j] = lam[j]; used[j] = 0; }
        for (int a = 0; a < NEV; a++) {
            int best = 0; float bv = -1e30f;
            for (int j = 0; j < NEV; j++)
                if (!used[j] && l[j] > bv) { bv = l[j]; best = j; }
            used[best] = 1; ord[a] = best;
        }
        for (int a = 0; a < NEV; a++) lam[a] = l[ord[a]];
    }
    __syncthreads();
    int i = threadIdx.x;
    float tmp[NEV];
    for (int a = 0; a < NEV; a++) tmp[a] = Vm[i * NEV + ord[a]];
    for (int a = 0; a < NEV; a++) Vm[i * NEV + a] = tmp[a];
}

// ---------------- per-d isomap post: xp, pip, logmap ----------------

__global__ void k_iso_post(const float* Vm, const float* lam, const float* X,
                           int d, float* xl, float* pip_out) {
    __shared__ float red[512];
    int i = threadIdx.x;
    float xp[6];
    for (int j = 0; j < d; j++) xp[j] = Vm[i * NEV + j] * sqrtf(fmaxf(lam[j], 0.f));
    float col = X[(size_t)i * E_N + (E_N - d)];
    float pip = 0.f;
    for (int j = 0; j < d; j++) {
        float dot = blk_reduce(xp[j] * col, red);
        pip += dot * dot;
    }
    float n2 = 0.f;
    for (int j = 0; j < d; j++) n2 += xp[j] * xp[j];
    float inv = rsqrtf(fmaxf(n2, 1e-30f));
    float xn[6];
    for (int j = 0; j < d; j++) xn[j] = xp[j] * inv;
    float last = fminf(fmaxf(xn[d - 1], -1.f + 1e-6f), 1.f - 1e-6f);
    float theta = acosf(last);
    float mean = 0.f;
    for (int j = 0; j < d; j++) mean += xn[j];
    mean /= (float)d;
    float scale = theta / sinf(theta);
    for (int j = 0; j < d - 1; j++) xl[i * (d - 1) + j] = (xn[j] - mean) * scale;
    if (i == 0) pip_out[0] = 0.001f * (float)(E_N - d) + 2.f * pip;
}

// ---------------- LDA ----------------

__global__ void k_counts(const int* T, float* counts, int n) {
    int i = blockIdx.x * blockDim.x + threadIdx.x;
    if (i < n) atomicAdd(&counts[T[i]], 1.f);
}

__global__ void k_classsum(const float* xl, const int* T, float* sums, int n, int k) {
    int i = blockIdx.x * blockDim.x + threadIdx.x;
    if (i >= n) return;
    int c = T[i];
    for (int j = 0; j < k; j++) atomicAdd(&sums[c * k + j], xl[i * k + j]);
}

__global__ void k_overall(const float* xl, int n, int k, float* ov) {
    __shared__ float red[512];
    int i = threadIdx.x;
    for (int j = 0; j < k; j++) {
        float s = blk_reduce(xl[i * k + j], red);
        if (i == 0) ov[j] = s / n;
    }
}

__global__ void k_sw(const float* xl, const int* T, const float* sums,
                     const float* counts, int n, int k, float* swm) {
    __shared__ float red[512];
    int i = threadIdx.x;
    int c = T[i];
    float cnt = fmaxf(counts[c], 1.f);
    float diff[5];
    for (int j = 0; j < k; j++) diff[j] = xl[i * k + j] - sums[c * k + j] / cnt;
    for (int a = 0; a < k; a++)
        for (int b = 0; b < k; b++) {
            float s = blk_reduce(diff[a] * diff[b], red);
            if (i == 0) swm[a * k + b] = s;
        }
}

__global__ void k_sb(const float* sums, const float* counts, const float* ov,
                     int nc, int k, float* sbm) {
    __shared__ float red[512];
    int c = threadIdx.x;
    float cnt = counts[c];
    float mask = (cnt > 0.f) ? 1.f : 0.f;
    float dd[5];
    for (int j = 0; j < k; j++) dd[j] = (sums[c * k + j] / fmaxf(cnt, 1.f) - ov[j]) * mask;
    for (int a = 0; a < k; a++)
        for (int b = 0; b < k; b++) {
            float s = blk_reduce(dd[a] * dd[b], red);
            if (c == 0) sbm[a * k + b] = 4.f * s;
        }
}

// fisher = (Sw + 1e-3 I)^-1 Sb ; eig-mean = trace/k ; dominant eigvec by power iter
__global__ void k_fisher(const float* swm, const float* sbm, int k,
                         float* evec, float* eigmean_out, float* meanlog_out) {
    float A[5][5], F[5][5];
    for (int a = 0; a < k; a++)
        for (int b = 0; b < k; b++) {
            A[a][b] = swm[a * k + b] + ((a == b) ? 1e-3f : 0.f);
            F[a][b] = sbm[a * k + b];
        }
    for (int p = 0; p < k; p++) {
        int pr = p; float bv = fabsf(A[p][p]);
        for (int r2 = p + 1; r2 < k; r2++)
            if (fabsf(A[r2][p]) > bv) { bv = fabsf(A[r2][p]); pr = r2; }
        if (pr != p) {
            for (int c = 0; c < k; c++) {
                float t = A[p][c]; A[p][c] = A[pr][c]; A[pr][c] = t;
                t = F[p][c]; F[p][c] = F[pr][c]; F[pr][c] = t;
            }
        }
        float piv = A[p][p];
        if (fabsf(piv) < 1e-20f) piv = 1e-20f;
        float ip = 1.f / piv;
        for (int c = 0; c < k; c++) { A[p][c] *= ip; F[p][c] *= ip; }
        for (int r2 = 0; r2 < k; r2++) {
            if (r2 == p) continue;
            float f = A[r2][p];
            for (int c = 0; c < k; c++) { A[r2][c] -= f * A[p][c]; F[r2][c] -= f * F[p][c]; }
        }
    }
    float tr = 0.f;
    for (int j = 0; j < k; j++) tr += F[j][j];
    float em = tr / (float)k;
    eigmean_out[0] = em;
    meanlog_out[0] = 0.1f * fabsf(em);
    float v[5];
    for (int j = 0; j < k; j++) v[j] = rsqrtf((float)k);
    for (int it = 0; it < 64; it++) {
        float nv[5];
        float nn = 0.f;
        for (int a = 0; a < k; a++) {
            float s = 0.f;
            for (int b = 0; b < k; b++) s += F[a][b] * v[b];
            nv[a] = s; nn += s * s;
        }
        nn = sqrtf(nn);
        if (nn < 1e-20f) nn = 1.f;
        for (int a = 0; a < k; a++) v[a] = nv[a] / nn;
    }
    for (int j = 0; j < k; j++) evec[j] = v[j];
}

__global__ void k_ref(const float* u, const float* evec, const float* center_p,
                      const float* meanlog_p, int k, int n, float* ref_out) {
    __shared__ float red[512];
    int i = threadIdx.x;
    float center = center_p[0], ml = meanlog_p[0];
    float s = 0.f;
    for (int j = 0; j < k; j++)
        s += (center - ml + u[i * 5 + j] * (2.f * ml)) * evec[j];
    float m = blk_reduce(fabsf(s), red) / n;
    if (i == 0) {
        float r = sqrtf(m);
        ref_out[0] = fminf(fmaxf(r, 0.7f), 1.5f);
    }
}

// ---------------- final proxy-anchor terms ----------------

__global__ void k_center(const float* cosm, int n2, float* out) {
    __shared__ float red[1024];
    float s = 0.f;
    for (int idx = threadIdx.x; idx < n2; idx += blockDim.x) s += cosm[idx];
    red[threadIdx.x] = s; __syncthreads();
    for (int o = blockDim.x >> 1; o > 0; o >>= 1) {
        if (threadIdx.x < o) red[threadIdx.x] += red[threadIdx.x + o];
        __syncthreads();
    }
    if (threadIdx.x == 0) out[0] = fabsf(red[0] / n2) + OFFSET_F;
}

__global__ void k_numvalid(const float* counts, int nc, float* out) {
    __shared__ float red[512];
    float v = (counts[threadIdx.x] > 0.f) ? 1.f : 0.f;
    float s = blk_reduce(v, red);
    if (threadIdx.x == 0) out[0] = s;
}

__global__ void k_select(float* sc) {
    int m = 0; float bv = sc[3];
    if (sc[4] < bv) { bv = sc[4]; m = 1; }
    if (sc[5] < bv) { bv = sc[5]; m = 2; }
    sc[12] = sc[6 + m];
    sc[13] = sc[9 + m];
}

__global__ void k_simsums(const float* cosm, const int* T, const float* ref_p,
                          float* pos_acc, float* neg_acc, int n, int nc) {
    int c = blockIdx.x * blockDim.x + threadIdx.x;
    if (c >= nc) return;
    float ref = ref_p[0];
    float ps = 0.f, ns = 0.f;
    for (int i = 0; i < n; i++) {
        float cs = cosm[(size_t)i * nc + c];
        if (T[i] == c) ps += __expf(-48.f * (cs - 0.1f));
        else           ns += __expf(48.f * (cs + 0.1f));
    }
    atomicAdd(pos_acc, logf(1.f + ps * ref));
    atomicAdd(neg_acc, logf(1.f + ns * ref));
}

__global__ void k_final(const float* sc, float* out) {
    out[0] = sc[14] / fmaxf(sc[16], 1.f) + sc[15] / (float)C_N - 1e-6f * sc[13];
}

// ---------------- host orchestration ----------------

extern "C" void kernel_launch(void* const* d_in, const int* in_sizes, int n_in,
                              void* d_out, int out_size, void* d_ws, size_t ws_size,
                              hipStream_t stream) {
    (void)in_sizes; (void)n_in; (void)out_size; (void)ws_size;
    const float* X       = (const float*)d_in[0];
    const int*   T       = (const int*)d_in[1];
    const float* u       = (const float*)d_in[2];
    const float* proxies = (const float*)d_in[3];

    char* base = (char*)d_ws;
    size_t off = 0;
    auto ALLOC = [&](size_t bytes) -> char* {
        char* p = base + off;
        off = (off + bytes + 255) & ~(size_t)255;
        return p;
    };

    _Float16* Xh_n = (_Float16*)ALLOC((size_t)B_N * E_N * 2);
    _Float16* Ph_n = (_Float16*)ALLOC((size_t)C_N * E_N * 2);
    _Float16* Xh_r = (_Float16*)ALLOC((size_t)B_N * E_N * 2);
    float* sq    = (float*)ALLOC(B_N * 4);
    float* cosm  = (float*)ALLOC((size_t)B_N * C_N * 4);
    float* gram  = (float*)ALLOC((size_t)B_N * B_N * 4);
    float* dist  = (float*)ALLOC((size_t)B_N * B_N * 4);
    float* g     = (float*)ALLOC((size_t)B_N * B_N * 4);
    float* kmat  = (float*)ALLOC((size_t)B_N * B_N * 4);
    float* rm    = (float*)ALLOC(B_N * 4);
    float* V     = (float*)ALLOC(B_N * NEV * 4);
    float* W     = (float*)ALLOC(B_N * NEV * 4);
    float* lam   = (float*)ALLOC(NEV * 4);
    float* xl    = (float*)ALLOC(B_N * 8 * 4);
    float* counts= (float*)ALLOC(C_N * 4);
    float* sums  = (float*)ALLOC(C_N * 8 * 4);
    float* ov    = (float*)ALLOC(8 * 4);
    float* swm   = (float*)ALLOC(32 * 4);
    float* sbm   = (float*)ALLOC(32 * 4);
    float* evec  = (float*)ALLOC(8 * 4);
    float* sc    = (float*)ALLOC(64 * 4);

    // stage: norms + f16 copies
    k_rownorm<<<B_N, 256, 0, stream>>>(X, E_N, sq, Xh_r, Xh_n);
    k_rownorm<<<C_N, 256, 0, stream>>>(proxies, E_N, nullptr, nullptr, Ph_n);

    // cos = Xn @ Pn^T ; gram = X @ X^T (WMMA + TDM-staged B panel)
    k_gemm16<<<(C_N / 16) * (B_N / 128), 256, 0, stream>>>(Xh_n, Ph_n, cosm, B_N, C_N, E_N);
    k_gemm16<<<(B_N / 16) * (B_N / 128), 256, 0, stream>>>(Xh_r, Xh_r, gram, B_N, B_N, E_N);

    // isomap graph + Floyd-Warshall
    k_dist<<<(B_N * B_N) / 256, 256, 0, stream>>>(gram, sq, dist, B_N);
    k_fillf<<<(B_N * B_N) / 256, 256, 0, stream>>>(g, B_N * B_N, INF_F);
    k_knn<<<(B_N + 255) / 256, 256, 0, stream>>>(dist, g, B_N);
    k_sym<<<(B_N * B_N) / 256, 256, 0, stream>>>(g, B_N);
    for (int k = 0; k < B_N; k++)
        k_fw_step<<<(B_N * B_N) / 256, 256, 0, stream>>>(g, B_N, k);
    k_gmax<<<1, 1024, 0, stream>>>(g, B_N * B_N, sc + 0);
    k_gfix<<<(B_N * B_N) / 256, 256, 0, stream>>>(g, B_N * B_N, sc + 0);

    // centered kernel matrix
    k_kmrow<<<B_N, 256, 0, stream>>>(g, B_N, rm);
    k_meanvec<<<1, 512, 0, stream>>>(rm, B_N, sc + 1);
    k_kmat<<<(B_N * B_N) / 256, 256, 0, stream>>>(g, rm, sc + 1, kmat, B_N);

    // top-NEV eigenpairs via subspace iteration (shared across all LDS_DIMS)
    k_vinit<<<(B_N * NEV) / 256, 256, 0, stream>>>(V, B_N, NEV);
    for (int t = 0; t < EIG_IT; t++) {
        k_gemmKV<<<(B_N * NEV) / 256, 256, 0, stream>>>(kmat, V, W, B_N);
        k_mgs<<<1, 512, 0, stream>>>(W, V, B_N);
    }
    k_gemmKV<<<(B_N * NEV) / 256, 256, 0, stream>>>(kmat, V, W, B_N);
    k_rayleigh<<<1, 512, 0, stream>>>(V, W, lam, B_N);
    k_sortcols<<<1, 512, 0, stream>>>(V, lam, B_N);

    // scalars shared by all d
    k_center<<<1, 1024, 0, stream>>>(cosm, B_N * C_N, sc + 2);
    k_fillf<<<(C_N + 255) / 256, 256, 0, stream>>>(counts, C_N, 0.f);
    k_counts<<<(B_N + 255) / 256, 256, 0, stream>>>(T, counts, B_N);

    // per-d pipeline: d in {4,5,6}
    for (int di = 0; di < 3; di++) {
        int d = 4 + di;
        int k = d - 1;
        k_iso_post<<<1, 512, 0, stream>>>(V, lam, X, d, xl, sc + 3 + di);
        k_fillf<<<(C_N * 8) / 256, 256, 0, stream>>>(sums, C_N * k, 0.f);
        k_classsum<<<(B_N + 255) / 256, 256, 0, stream>>>(xl, T, sums, B_N, k);
        k_overall<<<1, 512, 0, stream>>>(xl, B_N, k, ov);
        k_sw<<<1, 512, 0, stream>>>(xl, T, sums, counts, B_N, k, swm);
        k_sb<<<1, 512, 0, stream>>>(sums, counts, ov, C_N, k, sbm);
        k_fisher<<<1, 1, 0, stream>>>(swm, sbm, k, evec, sc + 9 + di, sc + 23);
        k_ref<<<1, 512, 0, stream>>>(u, evec, sc + 2, sc + 23, k, B_N, sc + 6 + di);
    }

    // selection + proxy-anchor loss
    k_select<<<1, 1, 0, stream>>>(sc);
    k_fillf<<<1, 32, 0, stream>>>(sc + 14, 2, 0.f);
    k_simsums<<<(C_N + 255) / 256, 256, 0, stream>>>(cosm, T, sc + 12, sc + 14, sc + 15, B_N, C_N);
    k_numvalid<<<1, 512, 0, stream>>>(counts, C_N, sc + 16);
    k_final<<<1, 1, 0, stream>>>(sc, (float*)d_out);
}
